// PointerChain_9088150798675
// MI455X (gfx1250) — matrix operand order repaired and checked by reading, not compiled, not run on hardware
//
#include <hip/hip_runtime.h>

typedef __attribute__((ext_vector_type(16))) _Float16 v16h;
typedef __attribute__((ext_vector_type(8)))  float    v8f;
typedef __attribute__((ext_vector_type(4)))  float    v4f;

// ds_swizzle group-of-32: offset = {xor[14:10], or[9:5], and[4:0]}
#define SWZ(x, pat) __int_as_float(__builtin_amdgcn_ds_swizzle(__float_as_int(x), (pat)))

namespace {
constexpr int   kD    = 512;
constexpr int   kN    = 8192;
constexpr int   kB    = 8;
constexpr int   kRows = kB * kN;      // 65536
constexpr int   kHops = 3;
constexpr float kEps  = 1e-5f;
constexpr int   kRowsPerBlock = 128;  // 8 waves x 16 rows
}

__device__ __forceinline__ float hsum(v4f v)  { return (v.x + v.y) + (v.z + v.w); }
__device__ __forceinline__ float hsum2(v4f v) { return (v.x*v.x + v.y*v.y) + (v.z*v.z + v.w*v.w); }

__global__ __launch_bounds__(256) void pointer_chain_kernel(
    const float* __restrict__ h, const int* __restrict__ ptr0,
    const float* __restrict__ gamma, const float* __restrict__ beta,
    const float* __restrict__ projw, const float* __restrict__ projb,
    float* __restrict__ out)
{
    __shared__ __align__(16) float gwLds[kD];   // gamma*w table for current hop
    __shared__ float rowRes[8][16];             // per-wave S_xgw per row

    const int tid  = threadIdx.x;
    const int lane = tid & 31;
    const int wave = tid >> 5;
    const int hi   = lane >> 4;                 // lane-half (K split of WMMA layouts)
    const int r    = lane & 15;                 // row this lane owns (M index)
    const int rowBase = blockIdx.x * kRowsPerBlock;

    // chain[0]: passthrough of first_hop_ptr
    if (tid < kRowsPerBlock)
        out[rowBase + tid] = (float)ptr0[rowBase + tid];

    const int    gRow = rowBase + wave * 16 + r;          // same for lanes r and r+16
    const float* hB   = h + (size_t)(gRow >> 13) * (size_t)kN * (size_t)kD; // batch base

    float cur = (float)ptr0[gRow];

    for (int hop = 1; hop < kHops; ++hop) {
        const float* gv = gamma + hop * kD;
        const float* bv = beta  + hop * kD;
        const float* wv = projw + hop * kD;

        for (int i = tid; i < kD; i += 256)
            gwLds[i] = gv[i] * wv[i];
        __syncthreads();

        // per-hop scalars: S_gw = sum(g*w), S_bw = sum(beta*w), wave allreduce
        float sgw = 0.f, sbw = 0.f;
        for (int k = lane; k < kD; k += 32) {
            float w = wv[k];
            sgw += gv[k] * w;
            sbw += bv[k] * w;
        }
        sgw += SWZ(sgw, 0x401F); sbw += SWZ(sbw, 0x401F);   // xor 16
        sgw += SWZ(sgw, 0x201F); sbw += SWZ(sbw, 0x201F);   // xor 8
        sgw += SWZ(sgw, 0x101F); sbw += SWZ(sbw, 0x101F);   // xor 4
        sgw += SWZ(sgw, 0x081F); sbw += SWZ(sbw, 0x081F);   // xor 2
        sgw += SWZ(sgw, 0x041F); sbw += SWZ(sbw, 0x041F);   // xor 1

        int p = (int)cur;
        p = p < 0 ? 0 : (p > kN - 1 ? kN - 1 : p);
        const float* row = hB + (size_t)p * kD;             // gathered row for M=r

        float sx = 0.f, sxx = 0.f;
        v8f  c = {};
        const int q0 = hi * 8;    // A: 16-bit layout — low lanes K=0..7/16..23, high +8
        const int b0 = hi * 8;    // B: same 16-bit layout as A — low lanes K=0..7/16..23, high +8
#pragma unroll
        for (int ch = 0; ch < 16; ++ch) {
            const int kb = ch * 32;
            v4f f0 = *(const v4f*)(row + kb + q0);
            v4f f1 = *(const v4f*)(row + kb + q0 + 4);
            v4f f2 = *(const v4f*)(row + kb + q0 + 16);
            v4f f3 = *(const v4f*)(row + kb + q0 + 20);
            sx  += hsum(f0)  + hsum(f1)  + hsum(f2)  + hsum(f3);
            sxx += hsum2(f0) + hsum2(f1) + hsum2(f2) + hsum2(f3);

            v16h a;
            a[0] =(_Float16)f0.x; a[1] =(_Float16)f0.y; a[2] =(_Float16)f0.z; a[3] =(_Float16)f0.w;
            a[4] =(_Float16)f1.x; a[5] =(_Float16)f1.y; a[6] =(_Float16)f1.z; a[7] =(_Float16)f1.w;
            a[8] =(_Float16)f2.x; a[9] =(_Float16)f2.y; a[10]=(_Float16)f2.z; a[11]=(_Float16)f2.w;
            a[12]=(_Float16)f3.x; a[13]=(_Float16)f3.y; a[14]=(_Float16)f3.z; a[15]=(_Float16)f3.w;

            v4f g0 = *(const v4f*)(gwLds + kb + b0);
            v4f g1 = *(const v4f*)(gwLds + kb + b0 + 4);
            v4f g2 = *(const v4f*)(gwLds + kb + b0 + 16);
            v4f g3 = *(const v4f*)(gwLds + kb + b0 + 20);
            v16h bm;
            bm[0] =(_Float16)g0.x; bm[1] =(_Float16)g0.y; bm[2] =(_Float16)g0.z; bm[3] =(_Float16)g0.w;
            bm[4] =(_Float16)g1.x; bm[5] =(_Float16)g1.y; bm[6] =(_Float16)g1.z; bm[7] =(_Float16)g1.w;
            bm[8] =(_Float16)g2.x; bm[9] =(_Float16)g2.y; bm[10]=(_Float16)g2.z; bm[11]=(_Float16)g2.w;
            bm[12]=(_Float16)g3.x; bm[13]=(_Float16)g3.y; bm[14]=(_Float16)g3.z; bm[15]=(_Float16)g3.w;

            // D[m][n] += sum_k A[m][k]*gw[k]  (all columns identical by construction)
            c = __builtin_amdgcn_wmma_f32_16x16x32_f16(
                    false, a, false, bm, (short)0, c, false, false);
        }

        // combine K-halves of the LN statistics: row r lives on lanes r and r+16
        sx  += SWZ(sx,  0x401F);
        sxx += SWZ(sxx, 0x401F);

        // C/D layout: low lanes hold rows 0..7 in c[0..7], high lanes rows 8..15
        if (lane == 0) {
#pragma unroll
            for (int j = 0; j < 8; ++j) rowRes[wave][j] = c[j];
        }
        if (lane == 16) {
#pragma unroll
            for (int j = 0; j < 8; ++j) rowRes[wave][8 + j] = c[j];
        }
        asm volatile("s_wait_dscnt 0" ::: "memory");
        float sxgw = rowRes[wave][r];

        float mu    = sx  * (1.f / (float)kD);
        float var   = sxx * (1.f / (float)kD) - mu * mu;
        float rs    = rsqrtf(var + kEps);
        float logit = rs * (sxgw - mu * sgw) + sbw + projb[hop];
        float sig   = 1.f / (1.f + __expf(-logit));
        float nxt   = rintf(sig * (float)(kN - 1));

        if (lane < 16)
            out[(size_t)hop * (size_t)kRows + gRow] = nxt;
        cur = nxt;             // identical on both lane-halves -> no broadcast needed
        __syncthreads();       // protect gwLds refill on next hop
    }
}

extern "C" void kernel_launch(void* const* d_in, const int* in_sizes, int n_in,
                              void* d_out, int out_size, void* d_ws, size_t ws_size,
                              hipStream_t stream) {
    (void)in_sizes; (void)n_in; (void)out_size; (void)d_ws; (void)ws_size;
    const float* h     = (const float*)d_in[0];
    const int*   ptr0  = (const int*)d_in[1];
    const float* gamma = (const float*)d_in[2];
    const float* beta  = (const float*)d_in[3];
    const float* projw = (const float*)d_in[4];
    const float* projb = (const float*)d_in[5];
    float*       out   = (float*)d_out;

    dim3 grid(kRows / kRowsPerBlock);   // 512 blocks x 256 threads (8 waves)
    pointer_chain_kernel<<<grid, 256, 0, stream>>>(h, ptr0, gamma, beta, projw, projb, out);
}
